// Attention_11476152615407
// MI455X (gfx1250) — matrix operand order repaired and checked by reading, not compiled, not run on hardware
//
#include <hip/hip_runtime.h>
#include <hip/hip_bf16.h>
#include <math.h>

// ---------------------------------------------------------------------------
// ViT attention with Gaussian positional bias, bf16 WMMA pipeline for gfx1250.
// B=64, N=197 (pad 208), E=768, H=12, dh=64.
// ---------------------------------------------------------------------------

#define BB   64
#define NN   197
#define NP   208           // 13 * 16
#define NPV  224           // 7 * 32  (k-dim padding for the PV matmul)
#define EE   768
#define HH   12
#define DH   64
#define BH   (BB*HH)       // 768
#define EPSF 1e-6f

typedef __attribute__((ext_vector_type(16))) __bf16 v16bf;
typedef __attribute__((ext_vector_type(8)))  __bf16 v8bf;
typedef __attribute__((ext_vector_type(8)))  float  v8f;

__device__ __forceinline__ v8f wmma_bf16(v16bf a, v16bf b, v8f c) {
    // D = A(16x32 bf16) * B(32x16 bf16) + C(16x16 f32)
    return __builtin_amdgcn_wmma_f32_16x16x32_bf16(
        /*neg_a=*/false, a, /*neg_b=*/false, b,
        /*c_mod=*/(short)0, c, /*reuse_a=*/false, /*reuse_b=*/false);
}

// Assemble a 16-element A/B operand from two 16B-aligned 8-element chunks.
__device__ __forceinline__ v16bf ld2x8(const __bf16* p0, const __bf16* p1) {
    v8bf lo = *(const v8bf*)p0;
    v8bf hi = *(const v8bf*)p1;
    v16bf r;
#pragma unroll
    for (int i = 0; i < 8; ++i) { r[i] = lo[i]; r[8 + i] = hi[i]; }
    return r;
}

__device__ __forceinline__ float softplus_f(float x) {
    return (x > 15.f) ? x : __logf(1.f + __expf(x));
}

// ---------------------------------------------------------------------------
// Kernel 0a: x (f32, [B][197][768]) -> xb (bf16, [B][208][768], zero pad rows)
// ---------------------------------------------------------------------------
__global__ void k_cvt_x(const float* __restrict__ x, __bf16* __restrict__ xb) {
    size_t i = (size_t)blockIdx.x * blockDim.x + threadIdx.x;
    size_t total = (size_t)BB * NP * EE;
    if (i >= total) return;
    int e = (int)(i % EE);
    int n = (int)((i / EE) % NP);
    int b = (int)(i / ((size_t)EE * NP));
    float v = (n < NN) ? x[((size_t)b * NN + n) * EE + e] : 0.f;
    xb[i] = (__bf16)v;
}

// ---------------------------------------------------------------------------
// Kernel 0b: Wq/Wk/Wv (f32, [768][768] as [f][e]) -> wb (bf16, 3x[768][768])
// ---------------------------------------------------------------------------
__global__ void k_cvt_w(const float* __restrict__ wq, const float* __restrict__ wk,
                        const float* __restrict__ wv, __bf16* __restrict__ wb) {
    size_t i = (size_t)blockIdx.x * blockDim.x + threadIdx.x;
    const size_t per = (size_t)EE * EE;
    if (i >= 3 * per) return;
    int s = (int)(i / per);
    size_t r = i % per;
    const float* src = (s == 0) ? wq : (s == 1) ? wk : wv;
    wb[i] = (__bf16)src[r];
}

// ---------------------------------------------------------------------------
// Kernel 0c: zero Vt buffer (bf16, [BH][64][224])
// ---------------------------------------------------------------------------
__global__ void k_zero_vt(__bf16* __restrict__ vt) {
    size_t i = (size_t)blockIdx.x * blockDim.x + threadIdx.x;
    size_t total = (size_t)BH * DH * NPV;
    if (i < total) vt[i] = (__bf16)0.0f;
}

// ---------------------------------------------------------------------------
// Kernel 1: QKV projection (bf16 WMMA, f32 accum) + fused var/alpha heads.
// One wave -> 16 token rows x 64 feature cols of one projection.
// Waves: b(64) x ntile(13) x proj(3) x ftile(12) = 29952 ; 4 waves / block.
// ---------------------------------------------------------------------------
__global__ void __launch_bounds__(128)
k_qkv(const __bf16* __restrict__ xb, const __bf16* __restrict__ wb,
      const float* __restrict__ Wvar, const float* __restrict__ bvar,
      const float* __restrict__ Walpha, const float* __restrict__ balpha,
      __bf16* __restrict__ qb, __bf16* __restrict__ kb,
      __bf16* __restrict__ vt, float* __restrict__ pr) {
    const int lane = threadIdx.x & 31;
    const int l16  = lane & 15;
    const int hi   = lane >> 4;
    const int wid  = blockIdx.x * (blockDim.x >> 5) + (threadIdx.x >> 5);

    const int ft =  wid % 12;          // feature tile (== head for this proj)
    const int p  = (wid / 12) % 3;     // 0=q 1=k 2=v
    const int nt = (wid / 36) % 13;    // token tile
    const int b  =  wid / (36 * 13);

    const __bf16* xrow = xb + ((size_t)b * NP + (nt * 16 + l16)) * EE;
    const __bf16* wrows[4];
#pragma unroll
    for (int dt = 0; dt < 4; ++dt) {
        int f = ft * 64 + dt * 16 + l16;
        wrows[dt] = wb + (size_t)p * (EE * EE) + (size_t)f * EE;
    }

    v8f acc[4];
#pragma unroll
    for (int dt = 0; dt < 4; ++dt)
#pragma unroll
        for (int r = 0; r < 8; ++r) acc[dt][r] = 0.f;

#pragma unroll
    for (int ec = 0; ec < EE / 32; ++ec) {
        const __bf16* xr = xrow + ec * 32;
        v16bf a = ld2x8(xr + hi * 16, xr + hi * 16 + 8);
#pragma unroll
        for (int dt = 0; dt < 4; ++dt) {
            v16bf bm = *(const v16bf*)(wrows[dt] + ec * 32 + hi * 16);
            acc[dt] = wmma_bf16(a, bm, acc[dt]);
        }
    }

    const int bh = b * HH + ft;

    if (p == 0) {
        // ---- write Q (bf16 [bh][n][d]) ----
#pragma unroll
        for (int dt = 0; dt < 4; ++dt) {
            int d = dt * 16 + l16;
#pragma unroll
            for (int r = 0; r < 8; ++r) {
                int n = nt * 16 + r + 8 * hi;
                qb[((size_t)bh * NP + n) * DH + d] = (__bf16)acc[dt][r];
            }
        }
        // ---- fused var/alpha heads on the f32 accumulators ----
        float wv0[4], wv1[4], wa[4];
#pragma unroll
        for (int dt = 0; dt < 4; ++dt) {
            int d = dt * 16 + l16;
            wv0[dt] = Wvar[d];
            wv1[dt] = Wvar[DH + d];
            wa[dt]  = Walpha[d];
        }
        float bv0 = bvar[0], bv1 = bvar[1], ba = balpha[0];
#pragma unroll
        for (int r = 0; r < 8; ++r) {
            float s0 = 0.f, s1 = 0.f, sa = 0.f;
#pragma unroll
            for (int dt = 0; dt < 4; ++dt) {
                s0 += acc[dt][r] * wv0[dt];
                s1 += acc[dt][r] * wv1[dt];
                sa += acc[dt][r] * wa[dt];
            }
#pragma unroll
            for (int m = 1; m < 16; m <<= 1) {
                s0 += __shfl_xor(s0, m, 32);
                s1 += __shfl_xor(s1, m, 32);
                sa += __shfl_xor(sa, m, 32);
            }
            if (l16 == 0) {
                int n = nt * 16 + r + 8 * hi;
                float vx = softplus_f(s0 + bv0) + 2.f * EPSF;  // var+eps then +eps
                float vy = softplus_f(s1 + bv1) + 2.f * EPSF;
                float al = softplus_f(sa + ba);
                *(float4*)(pr + ((size_t)bh * NP + n) * 4) =
                    make_float4(al, 1.f / vx, 1.f / vy, 0.f);
            }
        }
    } else if (p == 1) {
        // ---- write K (bf16 [bh][n][d]) ----
#pragma unroll
        for (int dt = 0; dt < 4; ++dt) {
            int d = dt * 16 + l16;
#pragma unroll
            for (int r = 0; r < 8; ++r) {
                int n = nt * 16 + r + 8 * hi;
                kb[((size_t)bh * NP + n) * DH + d] = (__bf16)acc[dt][r];
            }
        }
    } else {
        // ---- write V transposed (bf16 [bh][d][n]) ----
#pragma unroll
        for (int dt = 0; dt < 4; ++dt) {
            int d = dt * 16 + l16;
#pragma unroll
            for (int r = 0; r < 8; ++r) {
                int n = nt * 16 + r + 8 * hi;
                vt[((size_t)bh * DH + d) * NPV + n] = (__bf16)acc[dt][r];
            }
        }
    }
}

// ---------------------------------------------------------------------------
// Kernel 2: attention. One wave -> 16 query rows of one (b,h).
// Scores: 13 k-tiles x 2 WMMAs + Gaussian bias in VALU; softmax via shfl;
// P restaged through LDS (C-layout -> A-layout); O = P*V via 7x4 WMMAs.
// Waves: bh(768) x qtile(13) = 9984 ; 4 waves / block.
// ---------------------------------------------------------------------------
__global__ void __launch_bounds__(128)
k_attn(const __bf16* __restrict__ qb, const __bf16* __restrict__ kb,
       const __bf16* __restrict__ vt, const float* __restrict__ pr,
       float* __restrict__ out) {
    __shared__ __bf16 Pbuf[4][16][NPV];   // 28672 B

    const int lane = threadIdx.x & 31;
    const int l16  = lane & 15;
    const int hi   = lane >> 4;
    const int warp = threadIdx.x >> 5;
    const int wid  = blockIdx.x * (blockDim.x >> 5) + warp;

    const int qt = wid % 13;
    const int bh = wid / 13;
    const int b  = bh / HH;
    const int h  = bh % HH;
    const int q0 = qt * 16;

    // zero the LDS pad columns [208, 224)
#pragma unroll
    for (int i = 0; i < 8; ++i) {
        int flat = lane * 8 + i;                 // 256 entries
        Pbuf[warp][flat >> 4][NP + (flat & 15)] = (__bf16)0.0f;
    }

    // ---- load Q tile as two A operands (d 0..31, d 32..63) ----
    const __bf16* qp = qb + ((size_t)bh * NP + (q0 + l16)) * DH;
    v16bf a0 = ld2x8(qp + hi * 16,       qp + hi * 16 + 8);
    v16bf a1 = ld2x8(qp + 32 + hi * 16,  qp + 32 + hi * 16 + 8);

    // ---- per-row bias params (alpha, 1/varx, 1/vary) and grid coords ----
    float al[8], ivx[8], ivy[8];
    float qx[8], qy[8];
#pragma unroll
    for (int r = 0; r < 8; ++r) {
        int q = q0 + r + 8 * hi;
        if (q >= 1 && q < NN) {
            float4 t = *(const float4*)(pr + ((size_t)bh * NP + q) * 4);
            al[r] = t.x; ivx[r] = t.y; ivy[r] = t.z;
        } else {
            al[r] = 0.f; ivx[r] = 0.f; ivy[r] = 0.f;
        }
        int g = q - 1;
        qx[r] = (float)(g / 14);
        qy[r] = (float)(g % 14);
    }

    // ---- scores: 13 tiles of 16 columns ----
    v8f c[13];
    const float scale = 0.125f;  // 1/sqrt(64)
#pragma unroll
    for (int t = 0; t < 13; ++t) {
        int kcol = t * 16 + l16;
        const __bf16* kp = kb + ((size_t)bh * NP + kcol) * DH;
        v16bf b0 = *(const v16bf*)(kp + hi * 16);
        v16bf b1 = *(const v16bf*)(kp + 32 + hi * 16);
        v8f s;
#pragma unroll
        for (int r = 0; r < 8; ++r) s[r] = 0.f;
        s = wmma_bf16(a0, b0, s);
        s = wmma_bf16(a1, b1, s);

        bool kin = (kcol >= 1) && (kcol < NN);
        int g = kcol - 1;
        float kx = (float)(g / 14);
        float ky = (float)(g % 14);
#pragma unroll
        for (int r = 0; r < 8; ++r) {
            float add = 0.f;
            if (kin) {
                float dx = qx[r] - kx;
                float dy = qy[r] - ky;
                add = al[r] * __expf(-0.5f * (dx * dx * ivx[r] + dy * dy * ivy[r]));
            }
            float sc = s[r] * scale + add;
            if (kcol >= NN) sc = -1e30f;
            c[t][r] = sc;
        }
    }

    // ---- row softmax (rows live in one 16-lane half each) ----
    float mx[8];
#pragma unroll
    for (int r = 0; r < 8; ++r) mx[r] = -1e30f;
#pragma unroll
    for (int t = 0; t < 13; ++t)
#pragma unroll
        for (int r = 0; r < 8; ++r) mx[r] = fmaxf(mx[r], c[t][r]);
#pragma unroll
    for (int r = 0; r < 8; ++r)
#pragma unroll
        for (int m = 1; m < 16; m <<= 1) mx[r] = fmaxf(mx[r], __shfl_xor(mx[r], m, 32));

    float sum[8];
#pragma unroll
    for (int r = 0; r < 8; ++r) sum[r] = 0.f;
#pragma unroll
    for (int t = 0; t < 13; ++t) {
#pragma unroll
        for (int r = 0; r < 8; ++r) {
            float p = __expf(c[t][r] - mx[r]);
            sum[r] += p;
            Pbuf[warp][r + 8 * hi][t * 16 + l16] = (__bf16)p;
        }
    }
    float li[8];
#pragma unroll
    for (int r = 0; r < 8; ++r) {
#pragma unroll
        for (int m = 1; m < 16; m <<= 1) sum[r] += __shfl_xor(sum[r], m, 32);
        li[r] = 1.f / sum[r];
    }

    __syncthreads();

    // ---- O = P * V : 7 k-chunks of 32, 4 d-tiles of 16 ----
    v8f o[4];
#pragma unroll
    for (int dt = 0; dt < 4; ++dt)
#pragma unroll
        for (int r = 0; r < 8; ++r) o[dt][r] = 0.f;

    const __bf16* prow = &Pbuf[warp][l16][0];
#pragma unroll
    for (int cc = 0; cc < 7; ++cc) {
        v16bf ap = ld2x8(prow + cc * 32 + hi * 16, prow + cc * 32 + hi * 16 + 8);
#pragma unroll
        for (int dt = 0; dt < 4; ++dt) {
            int d = dt * 16 + l16;
            v16bf bv = *(const v16bf*)(vt + ((size_t)bh * DH + d) * NPV + cc * 32 + hi * 16);
            o[dt] = wmma_bf16(ap, bv, o[dt]);
        }
    }

    // ---- normalize and store out[b][n][h*64+d] (f32) ----
#pragma unroll
    for (int dt = 0; dt < 4; ++dt) {
        int d = dt * 16 + l16;
#pragma unroll
        for (int r = 0; r < 8; ++r) {
            int n = q0 + r + 8 * hi;
            if (n < NN)
                out[((size_t)b * NN + n) * EE + h * DH + d] = o[dt][r] * li[r];
        }
    }
}

// ---------------------------------------------------------------------------
// Launch
// ---------------------------------------------------------------------------
extern "C" void kernel_launch(void* const* d_in, const int* in_sizes, int n_in,
                              void* d_out, int out_size, void* d_ws, size_t ws_size,
                              hipStream_t stream) {
    const float* x      = (const float*)d_in[0];
    const float* Wq     = (const float*)d_in[1];
    const float* Wk     = (const float*)d_in[2];
    const float* Wv     = (const float*)d_in[3];
    const float* Wvar   = (const float*)d_in[4];
    const float* bvar   = (const float*)d_in[5];
    const float* Walpha = (const float*)d_in[6];
    const float* balpha = (const float*)d_in[7];
    // d_in[8] (diff) is recomputed on the fly.

    char* ws = (char*)d_ws;
    const size_t sz_xb = (size_t)BB * NP * EE * 2;       // 20,447,232
    const size_t sz_wb = (size_t)3 * EE * EE * 2;        //  3,538,944
    const size_t sz_qb = (size_t)BH * NP * DH * 2;       // 20,447,232
    const size_t sz_kb = sz_qb;
    const size_t sz_vt = (size_t)BH * DH * NPV * 2;      // 22,020,096

    __bf16* xb = (__bf16*)(ws);
    __bf16* wb = (__bf16*)(ws + sz_xb);
    __bf16* qb = (__bf16*)(ws + sz_xb + sz_wb);
    __bf16* kb = (__bf16*)(ws + sz_xb + sz_wb + sz_qb);
    __bf16* vt = (__bf16*)(ws + sz_xb + sz_wb + sz_qb + sz_kb);
    float*  pr = (float*) (ws + sz_xb + sz_wb + sz_qb + sz_kb + sz_vt);

    {   // x -> bf16 (padded)
        size_t total = (size_t)BB * NP * EE;
        k_cvt_x<<<(unsigned)((total + 255) / 256), 256, 0, stream>>>(x, xb);
    }
    {   // weights -> bf16
        size_t total = (size_t)3 * EE * EE;
        k_cvt_w<<<(unsigned)((total + 255) / 256), 256, 0, stream>>>(Wq, Wk, Wv, wb);
    }
    {   // zero Vt (pad columns must be 0)
        size_t total = (size_t)BH * DH * NPV;
        k_zero_vt<<<(unsigned)((total + 255) / 256), 256, 0, stream>>>(vt);
    }
    {   // QKV projection + var/alpha heads: 29952 waves / 4 per block
        k_qkv<<<7488, 128, 0, stream>>>(xb, wb, Wvar, bvar, Walpha, balpha,
                                        qb, kb, vt, pr);
    }
    {   // attention: 9984 waves / 4 per block
        k_attn<<<2496, 128, 0, stream>>>(qb, kb, vt, pr, (float*)d_out);
    }
}
